// TimeShiftedCrossModalAttention_2448131359281
// MI455X (gfx1250) — matrix operand order repaired and checked by reading, not compiled, not run on hardware
//
#include <hip/hip_runtime.h>
#include <hip/hip_bf16.h>

typedef __attribute__((ext_vector_type(16))) _Float16 v16h;
typedef __attribute__((ext_vector_type(8)))  _Float16 v8h;
typedef __attribute__((ext_vector_type(8)))  float    v8f;

#define BATCH   2
#define TDIM    2048
#define DMODEL  512
#define HEADS   8
#define DHEAD   64
#define MTOT    (BATCH * TDIM)   /* 4096 */
#define NEGBIG  (-1.0e30f)

/* ---------------- conversion kernels ---------------- */

__global__ void cvt_f32_f16(const float* __restrict__ src,
                            _Float16* __restrict__ dst, int n) {
  int i = blockIdx.x * blockDim.x + threadIdx.x;
  if (i < n) dst[i] = (_Float16)src[i];
}

/* src: (K,N) f32 row-major  ->  dst: (N,K) f16 row-major (i.e. W^T) */
__global__ void cvtT_f32_f16(const float* __restrict__ src,
                             _Float16* __restrict__ dst, int K, int N) {
  int i = blockIdx.x * blockDim.x + threadIdx.x;
  if (i < K * N) {
    int k = i / N, n = i % N;
    dst[(size_t)n * K + k] = (_Float16)src[i];
  }
}

/* ---------------- generic WMMA GEMM ----------------
 * One wave (32 threads) computes a 32x64 tile of C = A * B.
 * A : (M,K) f16 row-major
 * Bt: (N,K) f16 row-major  (= B^T, so B-fragments are contiguous)
 * MODE 0: C f16 row-major (M,N)
 * MODE 1: C f16 transposed (N,M)      [for V^T]
 * MODE 2: C f32 row-major (M,N) + bias[n]
 */
template<int MODE>
__global__ void gemm_wmma(const _Float16* __restrict__ A,
                          const _Float16* __restrict__ Bt,
                          void* __restrict__ Cout,
                          const float* __restrict__ bias,
                          int M, int N, int K) {
  const int mtile = blockIdx.x * 32;
  const int ntile = blockIdx.y * 64;
  const int lane  = threadIdx.x;
  const int nl    = lane & 15;   /* column within 16 (B/C) / row (A) */
  const int g     = lane >> 4;   /* lane-half */

  v8f acc[2][4] = {{{}, {}, {}, {}}, {{}, {}, {}, {}}};
  const _Float16* arow0 = A + (size_t)(mtile + nl) * K;
  const _Float16* arow1 = arow0 + (size_t)16 * K;

  for (int kb = 0; kb < K; kb += 32) {
    /* A fragments: lane holds one row; K = kb+16g .. +15 contiguous */
    v16h a0, a1;
    {
      v8h lo = *(const v8h*)(arow0 + kb + 16 * g);
      v8h hi = *(const v8h*)(arow0 + kb + 16 * g + 8);
      #pragma unroll
      for (int e = 0; e < 8; ++e) { a0[e] = lo[e]; a0[e + 8] = hi[e]; }
    }
    {
      v8h lo = *(const v8h*)(arow1 + kb + 16 * g);
      v8h hi = *(const v8h*)(arow1 + kb + 16 * g + 8);
      #pragma unroll
      for (int e = 0; e < 8; ++e) { a1[e] = lo[e]; a1[e + 8] = hi[e]; }
    }

    #pragma unroll
    for (int nt = 0; nt < 4; ++nt) {
      /* B fragment: lane = column (ntile+nt*16+nl), K = kb+16g .. +15 contiguous */
      v16h b = *(const v16h*)(Bt + (size_t)(ntile + nt * 16 + nl) * K + kb + 16 * g);
      acc[0][nt] = __builtin_amdgcn_wmma_f32_16x16x32_f16(
          false, a0, false, b, (short)0, acc[0][nt], false, false);
      acc[1][nt] = __builtin_amdgcn_wmma_f32_16x16x32_f16(
          false, a1, false, b, (short)0, acc[1][nt], false, false);
    }
  }

  #pragma unroll
  for (int ms = 0; ms < 2; ++ms) {
    const int mbase = mtile + ms * 16;
    if constexpr (MODE == 0) {
      _Float16* C = (_Float16*)Cout;
      #pragma unroll
      for (int nt = 0; nt < 4; ++nt)
        #pragma unroll
        for (int r = 0; r < 8; ++r)
          C[(size_t)(mbase + r + 8 * g) * N + ntile + nt * 16 + nl] =
              (_Float16)acc[ms][nt][r];
    } else if constexpr (MODE == 1) {
      _Float16* Ct = (_Float16*)Cout;   /* (N, M) */
      #pragma unroll
      for (int nt = 0; nt < 4; ++nt) {
        v8h pack;
        #pragma unroll
        for (int r = 0; r < 8; ++r) pack[r] = (_Float16)acc[ms][nt][r];
        *(v8h*)(Ct + (size_t)(ntile + nt * 16 + nl) * M + mbase + 8 * g) = pack;
      }
    } else {
      float* C = (float*)Cout;
      #pragma unroll
      for (int nt = 0; nt < 4; ++nt) {
        float bv = bias[ntile + nt * 16 + nl];
        #pragma unroll
        for (int r = 0; r < 8; ++r)
          C[(size_t)(mbase + r + 8 * g) * N + ntile + nt * 16 + nl] =
              acc[ms][nt][r] + bv;
      }
    }
  }
}

/* lag-weight select: lw[min(lag,5)] for lag >= 0, via cndmask chain */
__device__ __forceinline__ float lag_sel(int lag, float w0, float w1, float w2,
                                         float w3, float w4, float w5) {
  float w = w0;
  w = (lag >= 1) ? w1 : w;
  w = (lag >= 2) ? w2 : w;
  w = (lag >= 3) ? w3 : w;
  w = (lag >= 4) ? w4 : w;
  w = (lag >= 5) ? w5 : w;
  return w;
}

/* ---------------- flash attention (one wave per (b,h,16-query tile)) -------
 * Softmax row statistics are computed via an LDS transpose (row-per-lane),
 * needing only 2 cross-lane shuffles per 32-key chunk instead of 64.
 */
__global__ void attention_wmma(const _Float16* __restrict__ qh,  /* (MTOT,512) */
                               const _Float16* __restrict__ kh,  /* (MTOT,512) */
                               const _Float16* __restrict__ vt,  /* (512,MTOT) = V^T */
                               const float* __restrict__ lw,     /* (8,6) */
                               _Float16* __restrict__ ah)        /* (MTOT,512) */
{
  const int it   = blockIdx.x;            /* query tile, 0..127 */
  const int bh   = blockIdx.y;            /* 0..15 */
  const int b    = bh / HEADS;
  const int h    = bh % HEADS;
  const int hbase = h * DHEAD;
  const int lane = threadIdx.x;
  const int nl   = lane & 15;
  const int g    = lane >> 4;

  __shared__ __align__(16) float    Sld[16][36];   /* padded: bank-conflict-free */
  __shared__ __align__(16) _Float16 Plds[16][32];
  __shared__ float Astat[16];
  __shared__ float Lstat[16];

  const float w0 = lw[h * 6 + 0], w1 = lw[h * 6 + 1], w2 = lw[h * 6 + 2];
  const float w3 = lw[h * 6 + 3], w4 = lw[h * 6 + 4], w5 = lw[h * 6 + 5];

  /* Q A-fragments for the two 32-wide d-chunks of this head */
  v16h aq[2];
  {
    const _Float16* qrow = qh + (size_t)(b * TDIM + it * 16 + nl) * DMODEL + hbase;
    #pragma unroll
    for (int c = 0; c < 2; ++c) {
      v8h lo = *(const v8h*)(qrow + c * 32 + 16 * g);
      v8h hi = *(const v8h*)(qrow + c * 32 + 16 * g + 8);
      #pragma unroll
      for (int e = 0; e < 8; ++e) { aq[c][e] = lo[e]; aq[c][e + 8] = hi[e]; }
    }
  }

  v8f o[4] = {{}, {}, {}, {}};
  /* per-lane softmax state for row nl (replicated in both lane-halves) */
  float mrowL = -3.0e38f, lrowL = 0.0f;

  const int iBase = it * 16 + 8 * g;               /* row of C-element r is iBase+r */
  const int nch   = (it * 16 + 15) / 32 + 1;       /* causal chunk count */

  for (int chk = 0; chk < nch; ++chk) {
    const int jc = chk * 32;

    /* --- S = Q * K^T for 32 keys (two 16-wide n-subtiles) --- */
    v8f s[2] = {{}, {}};
    #pragma unroll
    for (int sx = 0; sx < 2; ++sx) {
      const _Float16* krow =
          kh + (size_t)(b * TDIM + jc + sx * 16 + nl) * DMODEL + hbase;
      #pragma unroll
      for (int c = 0; c < 2; ++c) {
        v16h bk = *(const v16h*)(krow + c * 32 + 16 * g);
        s[sx] = __builtin_amdgcn_wmma_f32_16x16x32_f16(
            false, aq[c], false, bk, (short)0, s[sx], false, false);
      }
    }

    /* --- scale + lag bias + causal mask; dump S tile to LDS (C-layout) --- */
    __syncthreads();   /* prior iteration's Plds/Astat reads done */
    #pragma unroll
    for (int r = 0; r < 8; ++r) {
      const int i  = iBase + r;
      const int l0 = i - (jc + nl);
      const int l1 = l0 - 16;
      float v0 = (l0 < 0) ? NEGBIG
                          : s[0][r] * 0.125f + lag_sel(l0, w0, w1, w2, w3, w4, w5);
      float v1 = (l1 < 0) ? NEGBIG
                          : s[1][r] * 0.125f + lag_sel(l1, w0, w1, w2, w3, w4, w5);
      Sld[8 * g + r][nl]      = v0;
      Sld[8 * g + r][16 + nl] = v1;
    }
    __syncthreads();

    /* --- row-per-lane softmax: lane owns row nl, cols 16g..16g+15 --- */
    float sv[16];
    {
      const float* srow = &Sld[nl][16 * g];
      #pragma unroll
      for (int c = 0; c < 16; ++c) sv[c] = srow[c];
    }
    float mx = sv[0];
    #pragma unroll
    for (int c = 1; c < 16; ++c) mx = fmaxf(mx, sv[c]);
    mx = fmaxf(mx, __shfl_xor(mx, 16, 32));          /* combine two halves */
    const float mn = fmaxf(mrowL, mx);
    const float al = __expf(mrowL - mn);
    float rs = 0.0f;
    v8h pk0, pk1;
    #pragma unroll
    for (int c = 0; c < 8; ++c) {
      float p = __expf(sv[c] - mn);      rs += p; pk0[c] = (_Float16)p;
    }
    #pragma unroll
    for (int c = 0; c < 8; ++c) {
      float p = __expf(sv[8 + c] - mn);  rs += p; pk1[c] = (_Float16)p;
    }
    rs += __shfl_xor(rs, 16, 32);
    lrowL = lrowL * al + rs;
    mrowL = mn;

    /* P directly in row-major layout that the A-fragment load wants */
    *(v8h*)&Plds[nl][16 * g]     = pk0;
    *(v8h*)&Plds[nl][16 * g + 8] = pk1;
    Astat[nl] = al;            /* both halves write identical value */
    __syncthreads();

    /* --- rescale running O with per-row alpha --- */
    #pragma unroll
    for (int r = 0; r < 8; ++r) {
      const float a8 = Astat[8 * g + r];
      #pragma unroll
      for (int nt = 0; nt < 4; ++nt) o[nt][r] *= a8;
    }

    /* --- P A-fragment --- */
    v16h ap;
    {
      v8h lo = *(const v8h*)&Plds[nl][16 * g];
      v8h hi = *(const v8h*)&Plds[nl][16 * g + 8];
      #pragma unroll
      for (int e = 0; e < 8; ++e) { ap[e] = lo[e]; ap[e + 8] = hi[e]; }
    }

    /* --- O += P * V  (V^T rows contiguous in t) --- */
    #pragma unroll
    for (int nt = 0; nt < 4; ++nt) {
      v16h bv = *(const v16h*)(vt + (size_t)(hbase + nt * 16 + nl) * MTOT +
                               b * TDIM + jc + 16 * g);
      o[nt] = __builtin_amdgcn_wmma_f32_16x16x32_f16(
          false, ap, false, bv, (short)0, o[nt], false, false);
    }
  }

  /* --- normalize and store (f16, row-major (MTOT,512)) --- */
  Lstat[nl] = lrowL;
  __syncthreads();
  float inv[8];
  #pragma unroll
  for (int r = 0; r < 8; ++r) inv[r] = 1.0f / Lstat[8 * g + r];
  #pragma unroll
  for (int nt = 0; nt < 4; ++nt)
    #pragma unroll
    for (int r = 0; r < 8; ++r)
      ah[(size_t)(b * TDIM + iBase + r) * DMODEL + hbase + nt * 16 + nl] =
          (_Float16)(o[nt][r] * inv[r]);
}

/* ---------------- host launcher ---------------- */

extern "C" void kernel_launch(void* const* d_in, const int* in_sizes, int n_in,
                              void* d_out, int out_size, void* d_ws, size_t ws_size,
                              hipStream_t stream) {
  const float* x   = (const float*)d_in[0];
  const float* ctx = (const float*)d_in[1];
  const float* Wq  = (const float*)d_in[2];
  const float* Wk  = (const float*)d_in[3];
  const float* Wv  = (const float*)d_in[4];
  const float* lw  = (const float*)d_in[5];
  const float* Wo  = (const float*)d_in[6];
  const float* bo  = (const float*)d_in[7];
  float* out = (float*)d_out;

  char* ws = (char*)d_ws;
  size_t off = 0;
  auto alloc = [&](size_t bytes) -> char* {
    char* p = ws + off;
    off += (bytes + 255) & ~(size_t)255;
    return p;
  };
  const size_t tokBytes = (size_t)MTOT * DMODEL * sizeof(_Float16);   /* 4 MiB */
  const size_t wBytes   = (size_t)DMODEL * DMODEL * sizeof(_Float16); /* 512 KiB */
  _Float16* xh  = (_Float16*)alloc(tokBytes);
  _Float16* chx = (_Float16*)alloc(tokBytes);
  _Float16* qh  = (_Float16*)alloc(tokBytes);
  _Float16* kh  = (_Float16*)alloc(tokBytes);
  _Float16* vtt = (_Float16*)alloc(tokBytes);   /* V^T (512, MTOT) */
  _Float16* ah  = (_Float16*)alloc(tokBytes);
  _Float16* Wqt = (_Float16*)alloc(wBytes);
  _Float16* Wkt = (_Float16*)alloc(wBytes);
  _Float16* Wvt = (_Float16*)alloc(wBytes);
  _Float16* Wot = (_Float16*)alloc(wBytes);
  (void)ws_size; (void)in_sizes; (void)n_in; (void)out_size;

  const int nTok = MTOT * DMODEL;       /* 2,097,152 */
  const int nW   = DMODEL * DMODEL;     /* 262,144 */
  cvt_f32_f16<<<(nTok + 255) / 256, 256, 0, stream>>>(x, xh, nTok);
  cvt_f32_f16<<<(nTok + 255) / 256, 256, 0, stream>>>(ctx, chx, nTok);
  cvtT_f32_f16<<<(nW + 255) / 256, 256, 0, stream>>>(Wq, Wqt, DMODEL, DMODEL);
  cvtT_f32_f16<<<(nW + 255) / 256, 256, 0, stream>>>(Wk, Wkt, DMODEL, DMODEL);
  cvtT_f32_f16<<<(nW + 255) / 256, 256, 0, stream>>>(Wv, Wvt, DMODEL, DMODEL);
  cvtT_f32_f16<<<(nW + 255) / 256, 256, 0, stream>>>(Wo, Wot, DMODEL, DMODEL);

  dim3 gp(MTOT / 32, DMODEL / 64);   /* (128, 8) waves, 32x64 C tiles */
  gemm_wmma<0><<<gp, 32, 0, stream>>>(xh,  Wqt, qh,  nullptr, MTOT, DMODEL, DMODEL);
  gemm_wmma<0><<<gp, 32, 0, stream>>>(chx, Wkt, kh,  nullptr, MTOT, DMODEL, DMODEL);
  gemm_wmma<1><<<gp, 32, 0, stream>>>(chx, Wvt, vtt, nullptr, MTOT, DMODEL, DMODEL);

  dim3 ga(TDIM / 16, BATCH * HEADS); /* (128, 16) waves */
  attention_wmma<<<ga, 32, 0, stream>>>(qh, kh, vtt, lw, ah);

  gemm_wmma<2><<<gp, 32, 0, stream>>>(ah, Wot, out, bo, MTOT, DMODEL, DMODEL);
}
